// PatchMerging3D_10505490006555
// MI455X (gfx1250) — matrix operand order repaired and checked by reading, not compiled, not run on hardware
//
#include <hip/hip_runtime.h>
#include <hip/hip_bf16.h>

typedef __attribute__((ext_vector_type(16))) __bf16 v16bf;
typedef __attribute__((ext_vector_type(8)))  __bf16 v8bf;
typedef __attribute__((ext_vector_type(8)))  float  v8f;
typedef __attribute__((ext_vector_type(4)))  unsigned v4u;
typedef __attribute__((ext_vector_type(8)))  int      v8i;
typedef __attribute__((ext_vector_type(4)))  int      v4i;

#if defined(__gfx1250__) && __has_builtin(__builtin_amdgcn_tensor_load_to_lds)
#define USE_TDM 1
#else
#define USE_TDM 0
#endif

// ---------------- problem constants ----------------
constexpr int DD = 40, HH = 400, WW = 352;
constexpr int NPTS = 200000, HIDN = 8;
constexpr int KSEL = 40000;                   // N * 0.2
constexpr int NTOT = NPTS + 4 * KSEL;         // 360000
constexpr int SZc  = DD / 2;                  // 20
constexpr int SYZc = SZc * (HH / 2);          // 4000
constexpr int SXYZc= SYZc * (WW / 2);         // 704000
constexpr int HWc  = HH * WW;                 // 140800
constexpr long DHWc = (long)DD * HWc;         // 5,632,000
constexpr int MVOX = SXYZc + 1;               // 704001 voxel-key space (+fill)
constexpr int SCAN_BLK = 1024;                // elements per scan block
constexpr int NB_SCAN  = (MVOX + SCAN_BLK - 1) / SCAN_BLK; // 688

// scalar slots (uint) in workspace
enum { SC_OWNMIN=0, SC_OWNMAX=1, SC_MXMIN=2, SC_MXMAX=3, SC_LOCMIN=4,
       SC_LOCMAX=5, SC_PREFIX=6, SC_REMAIN=7, SC_SELCNT=8, SC_EQCNT=9, SC_UCNT=10 };

// ---------------- helpers ----------------
__device__ __forceinline__ unsigned fenc(float f) {
    unsigned u = __float_as_uint(f);
    return (u & 0x80000000u) ? ~u : (u | 0x80000000u);   // monotone float->uint
}
__device__ __forceinline__ float fdec(unsigned u) {
    unsigned b = (u & 0x80000000u) ? (u & 0x7FFFFFFFu) : ~u;
    return __uint_as_float(b);
}
__device__ __forceinline__ v16bf cat16(v8bf lo, v8bf hi) {
    return __builtin_shufflevector(lo, hi, 0,1,2,3,4,5,6,7,8,9,10,11,12,13,14,15);
}

// ---------------- generic fills ----------------
__global__ void fillI32(int* p, int v, long n) {
    long i = (long)blockIdx.x * blockDim.x + threadIdx.x;
    if (i < n) p[i] = v;
}
__global__ void fillF32(float* p, float v, long n) {
    long i = (long)blockIdx.x * blockDim.x + threadIdx.x;
    if (i < n) p[i] = v;
}
__global__ void initScalK(unsigned* sc) {
    sc[SC_OWNMIN]=0xFFFFFFFFu; sc[SC_OWNMAX]=0u;
    sc[SC_MXMIN]=0xFFFFFFFFu;  sc[SC_MXMAX]=0u;
    sc[SC_LOCMIN]=0xFFFFFFFFu; sc[SC_LOCMAX]=0u;
    sc[SC_PREFIX]=0u; sc[SC_REMAIN]=(unsigned)KSEL;
    sc[SC_SELCNT]=0u; sc[SC_EQCNT]=0u; sc[SC_UCNT]=0u;
}

// ---------------- 1. dense point hash table ----------------
__global__ void buildPtabK(const int* __restrict__ idx, int* __restrict__ ptab) {
    int i = blockIdx.x * blockDim.x + threadIdx.x;
    if (i >= NPTS) return;
    int z = idx[i*4+1], y = idx[i*4+2], x = idx[i*4+3];
    ptab[z * HWc + y * WW + x] = i;
}

// ---------------- 1b. weight prep: transpose + bf16 convert ----------------
// wT[o][n][k] = bf16(conv_w[o][k][n])  (n-major so B fragments are contiguous)
__global__ void wprepK(const float* __restrict__ convw, __bf16* __restrict__ wT) {
    int i = blockIdx.x * blockDim.x + threadIdx.x;
    if (i >= 27 * 4096) return;
    int o = i >> 12, n = (i >> 6) & 63, k = i & 63;
    wT[i] = (__bf16)convw[o * 4096 + k * 64 + n];
}

// ---------------- 2. 27-tap sparse conv via WMMA bf16 ----------------
// block: 256 thr = 8 waves; wave w owns points [blockBase + 16w, +16)
__global__ __launch_bounds__(256) void convWmmaK(
        const float* __restrict__ feats, const __bf16* __restrict__ wT,
        const int* __restrict__ idx, const int* __restrict__ ptab,
        float* __restrict__ outFeat) {
    __shared__ __align__(16) __bf16 wtileT[64 * 64];    // n-major weights
    __shared__ __align__(16) __bf16 atile[8][16 * 64];  // per-wave gathered A tiles

    const int tid  = threadIdx.x;
    const int wave = tid >> 5, lane = tid & 31;
    const int blockBase = blockIdx.x * 128;

    // gather role: 2 lanes per row
    const int grow = lane >> 1, gpart = lane & 1;
    const int gpi  = blockBase + wave * 16 + grow;
    int pz = 0, py = 0, px = 0;
    const bool gvalid = (gpi < NPTS);
    if (gvalid) { pz = idx[gpi*4+1]; py = idx[gpi*4+2]; px = idx[gpi*4+3]; }

    // fragment role
    const int mrow = lane & 15, half = lane >> 4;

    v8f acc[4] = {};

    for (int o = 0; o < 27; ++o) {
        // ---- stage weight tile (8 KB contiguous) into LDS ----
#if USE_TDM
        if (wave == 0) {
            unsigned long ga = (unsigned long)(const void*)(wT + o * 4096);
            unsigned      la = (unsigned)(unsigned long)(const void*)wtileT;
            v4u g0 = {0, 0, 0, 0};
            g0[0] = 1u;                                  // count=1, user mode
            g0[1] = la;                                  // lds_addr
            g0[2] = (unsigned)ga;                        // global_addr[31:0]
            g0[3] = (unsigned)((ga >> 32) & 0x01FFFFFFul) | 0x80000000u; // ga[56:32] | type=2
            v8i g1 = {0, 0, 0, 0, 0, 0, 0, 0};
            g1[0] = 1 << 16;                             // data_size=2B, no multicast
            g1[1] = 4096 << 16;                          // tensor_dim0 = 4096 (lo16)
            g1[2] = 1 << 16;                             // dim0 hi=0, tensor_dim1 = 1 (lo16)
            g1[3] = 4096 << 16;                          // dim1 hi=0, tile_dim0 = 4096
            g1[4] = 1;                                   // tile_dim1 = 1, tile_dim2 = 0
            g1[5] = 4096;                                // tensor_dim0_stride lo32
            g1[6] = 0;                                   // stride0 hi, stride1 lo16
            g1[7] = 0;
            v4i gz4 = {0, 0, 0, 0};
            v8i gz8 = {0, 0, 0, 0, 0, 0, 0, 0};
            __builtin_amdgcn_tensor_load_to_lds(g0, g1, gz4, gz4, gz8, 0);
#if __has_builtin(__builtin_amdgcn_s_wait_tensorcnt)
            __builtin_amdgcn_s_wait_tensorcnt(0);
#else
            asm volatile("s_wait_tensorcnt 0x0" ::: "memory");
#endif
        }
#else
        {   // fallback: vectorized copy, 256 thr x 2 x 16B
            const int4* gsrc = (const int4*)(wT + o * 4096);
            int4* ldst = (int4*)wtileT;
            ldst[tid * 2]     = gsrc[tid * 2];
            ldst[tid * 2 + 1] = gsrc[tid * 2 + 1];
        }
#endif
        // ---- gather neighbor rows (bf16) into per-wave LDS tile ----
        const int dz = o / 9 - 1, dy = (o / 3) % 3 - 1, dx = o % 3 - 1;
        int j = -1;
        if (gvalid) {
            int zn = pz + dz, yn = py + dy, xn = px + dx;
            if (zn >= 0 && zn < DD && yn >= 0 && yn < HH && xn >= 0 && xn < WW)
                j = ptab[zn * HWc + yn * WW + xn];
        }
        const float4* src = (const float4*)(feats + (long)(j < 0 ? 0 : j) * 64);
        v8bf* dst = (v8bf*)&atile[wave][grow * 64 + gpart * 32];
        #pragma unroll
        for (int t = 0; t < 4; ++t) {
            float4 va = (j >= 0) ? src[gpart * 8 + 2*t]     : make_float4(0.f,0.f,0.f,0.f);
            float4 vb = (j >= 0) ? src[gpart * 8 + 2*t + 1] : make_float4(0.f,0.f,0.f,0.f);
            v8bf w8;
            w8[0]=(__bf16)va.x; w8[1]=(__bf16)va.y; w8[2]=(__bf16)va.z; w8[3]=(__bf16)va.w;
            w8[4]=(__bf16)vb.x; w8[5]=(__bf16)vb.y; w8[6]=(__bf16)vb.z; w8[7]=(__bf16)vb.w;
            dst[t] = w8;
        }
        __syncthreads();

        // ---- fragments: each v16bf = two aligned 16B LDS blocks ----
        const v8bf* arow = (const v8bf*)&atile[wave][mrow * 64];   // 8 chunks of 8 halves
        v16bf a0 = cat16(arow[2 * half],     arow[2 * half + 1]);  // K 16h..16h+15
        v16bf a1 = cat16(arow[4 + 2 * half], arow[4 + 2 * half + 1]); // +32
        #pragma unroll
        for (int nt = 0; nt < 4; ++nt) {
            const v8bf* brow = (const v8bf*)&wtileT[(nt * 16 + mrow) * 64];
            v16bf b0 = cat16(brow[2 * half],     brow[2 * half + 1]);   // K 16h..16h+15
            v16bf b1 = cat16(brow[4 + 2 * half], brow[4 + 2 * half + 1]); // +32
            acc[nt] = __builtin_amdgcn_wmma_f32_16x16x32_bf16(
                          false, a0, false, b0, (short)0, acc[nt], false, false);
            acc[nt] = __builtin_amdgcn_wmma_f32_16x16x32_bf16(
                          false, a1, false, b1, (short)0, acc[nt], false, false);
        }
        __syncthreads();   // before next offset overwrites LDS
    }

    // ---- store D (ISA 16x16 f32 C/D layout: vgpr r -> M = r + 8*half, N = lane&15)
    #pragma unroll
    for (int nt = 0; nt < 4; ++nt) {
        #pragma unroll
        for (int r = 0; r < 8; ++r) {
            int M = r + 8 * half;
            int prow = blockBase + wave * 16 + M;
            if (prow < NPTS)
                outFeat[(long)prow * 64 + nt * 16 + mrow] = acc[nt][r];
        }
    }
}

// ---------------- 3. LN1 + exact GELU, row mean/max + global min/max ----------------
__global__ void lnGeluK(float* __restrict__ feat,
                        const float* __restrict__ lnw, const float* __restrict__ lnb,
                        float* __restrict__ ownraw, float* __restrict__ mxraw,
                        unsigned* __restrict__ sc) {
    int i = blockIdx.x * blockDim.x + threadIdx.x;
    if (i >= NPTS) return;
    float v[64];
    float mu = 0.f;
    #pragma unroll
    for (int d = 0; d < 64; ++d) { v[d] = feat[(long)i*64 + d]; mu += v[d]; }
    mu *= (1.f / 64.f);
    float var = 0.f;
    #pragma unroll
    for (int d = 0; d < 64; ++d) { float t = v[d] - mu; var += t * t; }
    var *= (1.f / 64.f);
    float rs = rsqrtf(var + 1e-5f);
    float sum = 0.f, mx = -3.4e38f;
    #pragma unroll
    for (int d = 0; d < 64; ++d) {
        float x = (v[d] - mu) * rs * lnw[d] + lnb[d];
        float g = 0.5f * x * (1.f + erff(x * 0.70710678118654752f));
        sum += g; mx = fmaxf(mx, g);
        feat[(long)i*64 + d] = g;
    }
    float mean = sum * (1.f / 64.f);
    ownraw[i] = mean; mxraw[i] = mx;
    atomicMin(&sc[SC_OWNMIN], fenc(mean)); atomicMax(&sc[SC_OWNMAX], fenc(mean));
    atomicMin(&sc[SC_MXMIN],  fenc(mx));   atomicMax(&sc[SC_MXMAX],  fenc(mx));
}

// ---------------- 4. 26-neighbor response average ----------------
__global__ void localK(const int* __restrict__ idx, const int* __restrict__ ptab,
                       const float* __restrict__ ownraw, float* __restrict__ localpre,
                       unsigned* __restrict__ sc) {
    int i = blockIdx.x * blockDim.x + threadIdx.x;
    if (i >= NPTS) return;
    float mn = fdec(sc[SC_OWNMIN]), mxv = fdec(sc[SC_OWNMAX]);
    float rng = 1.f / (mxv - mn + 1e-6f);
    int z = idx[i*4+1], y = idx[i*4+2], x = idx[i*4+3];
    float ssum = 0.f; int cnt = 0;
    for (int o = 0; o < 27; ++o) {
        if (o == 13) continue;
        int zn = z + o/9 - 1, yn = y + (o/3)%3 - 1, xn = x + o%3 - 1;
        if (zn < 0 || zn >= DD || yn < 0 || yn >= HH || xn < 0 || xn >= WW) continue;
        int j = ptab[zn * HWc + yn * WW + xn];
        if (j >= 0) { ssum += (ownraw[j] - mn) * rng; ++cnt; }
    }
    float lp = ssum / ((float)cnt + 1e-6f);
    localpre[i] = lp;
    atomicMin(&sc[SC_LOCMIN], fenc(lp)); atomicMax(&sc[SC_LOCMAX], fenc(lp));
}

// ---------------- 5. gating MLP + softmax -> att ----------------
__global__ void gateK(const float* __restrict__ ownraw, const float* __restrict__ mxraw,
                      const float* __restrict__ localpre,
                      const float* __restrict__ w1, const float* __restrict__ b1,
                      const float* __restrict__ bng, const float* __restrict__ bnb,
                      const float* __restrict__ w2, const float* __restrict__ b2,
                      const float* __restrict__ alpha, const unsigned* __restrict__ sc,
                      float* __restrict__ att, unsigned* __restrict__ key) {
    int i = blockIdx.x * blockDim.x + threadIdx.x;
    if (i >= NPTS) return;
    float omn = fdec(sc[SC_OWNMIN]), omx = fdec(sc[SC_OWNMAX]);
    float mmn = fdec(sc[SC_MXMIN]),  mmx = fdec(sc[SC_MXMAX]);
    float lmn = fdec(sc[SC_LOCMIN]), lmx = fdec(sc[SC_LOCMAX]);
    float own = (ownraw[i]   - omn) / (omx - omn + 1e-6f);
    float mx  = (mxraw[i]    - mmn) / (mmx - mmn + 1e-6f);
    float loc = (localpre[i] - lmn) / (lmx - lmn + 1e-6f);
    const float scale = rsqrtf(1.f + 1e-5f);
    float h[HIDN];
    #pragma unroll
    for (int jH = 0; jH < HIDN; ++jH) {
        float a = own * w1[0*HIDN + jH] + mx * w1[1*HIDN + jH] + loc * w1[2*HIDN + jH] + b1[jH];
        a = a * scale * bng[jH] + bnb[jH];
        a = a * (fminf(fmaxf(a + 3.f, 0.f), 6.f) * (1.f / 6.f));   // hardswish gate
        h[jH] = a;
    }
    float lg[3];
    #pragma unroll
    for (int c = 0; c < 3; ++c) {
        float a = b2[c];
        #pragma unroll
        for (int jH = 0; jH < HIDN; ++jH) a += h[jH] * w2[jH*3 + c];
        lg[c] = a;
    }
    float mlg = fmaxf(lg[0], fmaxf(lg[1], lg[2]));
    float e0 = expf(lg[0]-mlg), e1 = expf(lg[1]-mlg), e2 = expf(lg[2]-mlg);
    float inv = 1.f / (e0 + e1 + e2);
    float a0 = alpha[0];
    float av = a0 * (e0*inv*own + e1*inv*mx + e2*inv*loc)
             + (1.f - a0) * ((own + mx + loc) * (1.f/3.f));
    att[i] = av;
    key[i] = fenc(av);
}

// ---------------- 6. top-K radix select ----------------
__global__ void histK(const unsigned* __restrict__ key, unsigned* __restrict__ hist,
                      const unsigned* __restrict__ sc, int byteIdx) {
    int i = blockIdx.x * blockDim.x + threadIdx.x;
    if (i >= NPTS) return;
    unsigned k = key[i];
    unsigned prefix = sc[SC_PREFIX];
    unsigned mask = (byteIdx == 3) ? 0u : (0xFFFFFFFFu << (8 * (byteIdx + 1)));
    if ((k & mask) == (prefix & mask))
        atomicAdd(&hist[(k >> (8 * byteIdx)) & 255u], 1u);
}
__global__ void pickK(unsigned* __restrict__ hist, unsigned* __restrict__ sc, int byteIdx) {
    unsigned rem = sc[SC_REMAIN], prefix = sc[SC_PREFIX];
    for (int d = 255; d >= 0; --d) {
        unsigned c = hist[d];
        if (rem <= c) { prefix |= ((unsigned)d) << (8 * byteIdx); break; }
        rem -= c;
    }
    sc[SC_PREFIX] = prefix; sc[SC_REMAIN] = rem;
    for (int d = 0; d < 256; ++d) hist[d] = 0u;
}
__global__ void selectK(const unsigned* __restrict__ key, int* __restrict__ toplist,
                        unsigned* __restrict__ sc) {
    int i = blockIdx.x * blockDim.x + threadIdx.x;
    if (i >= NPTS) return;
    unsigned T = sc[SC_PREFIX], need = sc[SC_REMAIN];
    unsigned k = key[i];
    if (k > T) {
        unsigned p = atomicAdd(&sc[SC_SELCNT], 1u);
        toplist[p] = i;
    } else if (k == T) {
        unsigned e = atomicAdd(&sc[SC_EQCNT], 1u);
        if (e < need) {
            unsigned p = atomicAdd(&sc[SC_SELCNT], 1u);
            toplist[p] = i;
        }
    }
}

// ---------------- 7. merge keys + dense unique ----------------
__global__ void mkK(const int* __restrict__ idx, const int* __restrict__ toplist,
                    int* __restrict__ mk, int* __restrict__ present) {
    int r = blockIdx.x * blockDim.x + threadIdx.x;
    if (r >= NTOT) return;
    int z, y, x;
    if (r < NPTS) {
        z = idx[r*4+1]; y = idx[r*4+2]; x = idx[r*4+3];
    } else {
        int s = (r - NPTS) / KSEL, t = (r - NPTS) % KSEL;
        int sel = toplist[t];
        z = idx[sel*4+1]; y = idx[sel*4+2]; x = idx[sel*4+3];
        int dy = (s < 2) ? 1 : -1;
        int dx = (s & 1) ? 1 : -1;
        z = min(max(z, 0), DD - 1);
        y = min(max(y + dy, 0), HH - 1);
        x = min(max(x + dx, 0), WW - 1);
    }
    int v = (x >> 1) * SYZc + (y >> 1) * SZc + (z >> 1);
    mk[r] = v;
    present[v] = 1;
}

__global__ __launch_bounds__(256) void scan1K(const int* __restrict__ present,
                                              int* __restrict__ rank, int* __restrict__ bsums) {
    __shared__ int s[256];
    int t = threadIdx.x;
    long base = (long)blockIdx.x * SCAN_BLK + (long)t * 4;
    int v0 = (base + 0 < MVOX) ? present[base + 0] : 0;
    int v1 = (base + 1 < MVOX) ? present[base + 1] : 0;
    int v2 = (base + 2 < MVOX) ? present[base + 2] : 0;
    int v3 = (base + 3 < MVOX) ? present[base + 3] : 0;
    int tsum = v0 + v1 + v2 + v3;
    s[t] = tsum; __syncthreads();
    for (int off = 1; off < 256; off <<= 1) {
        int add = (t >= off) ? s[t - off] : 0;
        __syncthreads();
        s[t] += add;
        __syncthreads();
    }
    int excl = s[t] - tsum;
    if (base + 0 < MVOX) rank[base + 0] = excl;
    if (base + 1 < MVOX) rank[base + 1] = excl + v0;
    if (base + 2 < MVOX) rank[base + 2] = excl + v0 + v1;
    if (base + 3 < MVOX) rank[base + 3] = excl + v0 + v1 + v2;
    if (t == 0) bsums[blockIdx.x] = s[255];
}
__global__ void scan2K(int* __restrict__ bsums, unsigned* __restrict__ sc) {
    int run = 0;
    for (int b = 0; b < NB_SCAN; ++b) { int t = bsums[b]; bsums[b] = run; run += t; }
    sc[SC_UCNT] = (unsigned)run;
}
__global__ void scan3K(int* __restrict__ rank, const int* __restrict__ bsums) {
    int t = threadIdx.x;
    long base = (long)blockIdx.x * SCAN_BLK + (long)t * 4;
    int add = bsums[blockIdx.x];
    #pragma unroll
    for (int j = 0; j < 4; ++j)
        if (base + j < MVOX) rank[base + j] += add;
}
__global__ void unqFillK(int* __restrict__ unq) {
    int r = blockIdx.x * blockDim.x + threadIdx.x;
    if (r < NTOT) unq[r] = SXYZc;   // fill_value
}
__global__ void unqScatterK(const int* __restrict__ present, const int* __restrict__ rank,
                            int* __restrict__ unq) {
    int v = blockIdx.x * blockDim.x + threadIdx.x;
    if (v < MVOX && present[v]) unq[rank[v]] = v;
}
__global__ void invK(int* __restrict__ mk, const int* __restrict__ rank,
                     float* __restrict__ out) {
    int r = blockIdx.x * blockDim.x + threadIdx.x;
    if (r >= NTOT) return;
    int iv = rank[mk[r]];
    mk[r] = iv;   // reuse buffer as inv for segment sum
    out[(size_t)NTOT * 64 + (size_t)NTOT * 4 + r] = (float)iv;
}
__global__ void segK(const float* __restrict__ feat, const int* __restrict__ inv,
                     float* __restrict__ out) {
    long t = (long)blockIdx.x * blockDim.x + threadIdx.x;
    if (t >= (long)NPTS * 64) return;
    int i = (int)(t >> 6), d = (int)(t & 63);
    atomicAdd(&out[(long)inv[i] * 64 + d], feat[(long)i * 64 + d]);
}
__global__ void finalK(float* __restrict__ out, const float* __restrict__ lnw,
                       const float* __restrict__ lnb, const int* __restrict__ unq) {
    int r = blockIdx.x * blockDim.x + threadIdx.x;
    if (r >= NTOT) return;
    float v[64]; float mu = 0.f;
    #pragma unroll
    for (int d = 0; d < 64; ++d) { v[d] = out[(long)r*64 + d]; mu += v[d]; }
    mu *= (1.f / 64.f);
    float var = 0.f;
    #pragma unroll
    for (int d = 0; d < 64; ++d) { float t = v[d] - mu; var += t * t; }
    var *= (1.f / 64.f);
    float rs = rsqrtf(var + 1e-5f);
    #pragma unroll
    for (int d = 0; d < 64; ++d)
        out[(long)r*64 + d] = (v[d] - mu) * rs * lnw[d] + lnb[d];
    int u = unq[r];
    int vb = u / SXYZc;
    int vx = (u % SXYZc) / SYZc;
    int vy = (u % SYZc) / SZc;
    int vz = u % SZc;
    size_t vo = (size_t)NTOT * 64 + (size_t)r * 4;
    out[vo + 0] = (float)vb; out[vo + 1] = (float)vz;
    out[vo + 2] = (float)vy; out[vo + 3] = (float)vx;
}

// ---------------- workspace layout (bytes, 256-aligned) ----------------
constexpr size_t OFF_PTAB   = 0;                                   // int[DHW]
constexpr size_t OFF_FEAT   = OFF_PTAB   + 22528000;               // float[N*64]
constexpr size_t OFF_OWN    = OFF_FEAT   + 51200000;               // float[N]
constexpr size_t OFF_MX     = OFF_OWN    + 800000;                 // float[N]
constexpr size_t OFF_LOC    = OFF_MX     + 800000;                 // float[N]
constexpr size_t OFF_ATT    = OFF_LOC    + 800000;                 // float[N]
constexpr size_t OFF_KEY    = OFF_ATT    + 800000;                 // uint[N]
constexpr size_t OFF_TOP    = OFF_KEY    + 800000;                 // int[K]
constexpr size_t OFF_MK     = OFF_TOP    + 160000;                 // int[NT]
constexpr size_t OFF_UNQ    = OFF_MK     + 1440000;                // int[NT]
constexpr size_t OFF_PRES   = OFF_UNQ    + 1440000;                // int[MVOX]
constexpr size_t OFF_RANK   = OFF_PRES   + 2816256;                // int[MVOX]
constexpr size_t OFF_BSUM   = OFF_RANK   + 2816256;                // int[1024]
constexpr size_t OFF_HIST   = OFF_BSUM   + 4096;                   // uint[256]
constexpr size_t OFF_SCAL   = OFF_HIST   + 1024;                   // uint[64]
constexpr size_t OFF_WT     = OFF_SCAL   + 256;                    // bf16[27*4096]

extern "C" void kernel_launch(void* const* d_in, const int* in_sizes, int n_in,
                              void* d_out, int out_size, void* d_ws, size_t ws_size,
                              hipStream_t stream) {
    (void)in_sizes; (void)n_in; (void)out_size; (void)ws_size;
    const float* features = (const float*)d_in[0];
    const float* convw    = (const float*)d_in[1];
    const float* ln1w     = (const float*)d_in[2];
    const float* ln1b     = (const float*)d_in[3];
    const float* w1       = (const float*)d_in[4];
    const float* b1       = (const float*)d_in[5];
    const float* bng      = (const float*)d_in[6];
    const float* bnb      = (const float*)d_in[7];
    const float* w2       = (const float*)d_in[8];
    const float* b2       = (const float*)d_in[9];
    const float* alpha    = (const float*)d_in[10];
    const float* ln2w     = (const float*)d_in[11];
    const float* ln2b     = (const float*)d_in[12];
    const int*   indices  = (const int*)d_in[13];
    float* out = (float*)d_out;

    char* ws = (char*)d_ws;
    int*      ptab   = (int*)(ws + OFF_PTAB);
    float*    feat   = (float*)(ws + OFF_FEAT);
    float*    ownraw = (float*)(ws + OFF_OWN);
    float*    mxraw  = (float*)(ws + OFF_MX);
    float*    locpre = (float*)(ws + OFF_LOC);
    float*    att    = (float*)(ws + OFF_ATT);
    unsigned* key    = (unsigned*)(ws + OFF_KEY);
    int*      toplist= (int*)(ws + OFF_TOP);
    int*      mk     = (int*)(ws + OFF_MK);
    int*      unq    = (int*)(ws + OFF_UNQ);
    int*      present= (int*)(ws + OFF_PRES);
    int*      rank   = (int*)(ws + OFF_RANK);
    int*      bsums  = (int*)(ws + OFF_BSUM);
    unsigned* hist   = (unsigned*)(ws + OFF_HIST);
    unsigned* scal   = (unsigned*)(ws + OFF_SCAL);
    __bf16*   wT     = (__bf16*)(ws + OFF_WT);

    auto blocks = [](long n, int tpb) { return (int)((n + tpb - 1) / tpb); };

    // init
    fillI32<<<blocks(DHWc, 256), 256, 0, stream>>>(ptab, -1, DHWc);
    fillI32<<<blocks(MVOX, 256), 256, 0, stream>>>(present, 0, MVOX);
    fillI32<<<1, 256, 0, stream>>>((int*)hist, 0, 256);
    fillF32<<<blocks((long)NTOT * 69, 256), 256, 0, stream>>>(out, 0.f, (long)NTOT * 69);
    initScalK<<<1, 1, 0, stream>>>(scal);

    // pipeline
    buildPtabK<<<blocks(NPTS, 256), 256, 0, stream>>>(indices, ptab);
    wprepK<<<blocks(27 * 4096, 256), 256, 0, stream>>>(convw, wT);
    convWmmaK<<<blocks(NPTS, 128), 256, 0, stream>>>(features, wT, indices, ptab, feat);
    lnGeluK<<<blocks(NPTS, 256), 256, 0, stream>>>(feat, ln1w, ln1b, ownraw, mxraw, scal);
    localK<<<blocks(NPTS, 256), 256, 0, stream>>>(indices, ptab, ownraw, locpre, scal);
    gateK<<<blocks(NPTS, 256), 256, 0, stream>>>(ownraw, mxraw, locpre, w1, b1, bng, bnb,
                                                 w2, b2, alpha, scal, att, key);
    // top-K radix select (MSB -> LSB)
    for (int byteIdx = 3; byteIdx >= 0; --byteIdx) {
        histK<<<blocks(NPTS, 256), 256, 0, stream>>>(key, hist, scal, byteIdx);
        pickK<<<1, 1, 0, stream>>>(hist, scal, byteIdx);
    }
    selectK<<<blocks(NPTS, 256), 256, 0, stream>>>(key, toplist, scal);

    // merge keys + dense unique + segment sum + LN2
    mkK<<<blocks(NTOT, 256), 256, 0, stream>>>(indices, toplist, mk, present);
    scan1K<<<NB_SCAN, 256, 0, stream>>>(present, rank, bsums);
    scan2K<<<1, 1, 0, stream>>>(bsums, scal);
    scan3K<<<NB_SCAN, 256, 0, stream>>>(rank, bsums);
    unqFillK<<<blocks(NTOT, 256), 256, 0, stream>>>(unq);
    unqScatterK<<<blocks(MVOX, 256), 256, 0, stream>>>(present, rank, unq);
    invK<<<blocks(NTOT, 256), 256, 0, stream>>>(mk, rank, out);
    segK<<<blocks((long)NPTS * 64, 256), 256, 0, stream>>>(feat, mk, out);
    finalK<<<blocks(NTOT, 256), 256, 0, stream>>>(out, ln2w, ln2b, unq);
}
